// TokenEmbeder_83717502534011
// MI455X (gfx1250) — matrix operand order repaired and checked by reading, not compiled, not run on hardware
//
#include <hip/hip_runtime.h>
#include <hip/hip_bf16.h>
#include <stdint.h>

// ---------------------------------------------------------------------------
// Problem constants (match reference)
// ---------------------------------------------------------------------------
#define BB    256      // batch
#define TCODE 200      // code seq len
#define TDESC 50       // desc seq len
#define EE    300      // embedding dim
#define EPAD  320      // E padded to multiple of 32
#define HH    512      // hidden
#define G4H   2048     // 4*H
#define KPAD  832      // EPAD + H  (fused gate-GEMM K)
#define BD    512      // combined desc batch (anchor 0..255, neg 256..511)
#define MROWS (BB*TCODE) // 51200 feat rows

typedef __attribute__((ext_vector_type(16))) __bf16 v16bf;
typedef __attribute__((ext_vector_type(8)))  __bf16 v8bf;
typedef __attribute__((ext_vector_type(8)))  float  v8f;

union AU { v16bf v; v8bf h[2]; };

static __device__ __forceinline__ v8f wmma_bf16(v16bf a, v16bf b, v8f c) {
  return __builtin_amdgcn_wmma_f32_16x16x32_bf16(false, a, false, b, (short)0, c,
                                                 false, false);
}

// Load one 16x32 bf16 A fragment for this lane directly from a row pointer.
// Lane (half, lm): row = lm, contiguous chunk [k0+half*16, +16).
static __device__ __forceinline__ v16bf load_afrag(const __hip_bfloat16* row,
                                                   int k0, int half) {
  AU a;
  a.h[0] = *(const v8bf*)(row + k0 + half * 16);
  a.h[1] = *(const v8bf*)(row + k0 + half * 16 + 8);
  return a.v;
}

static __device__ __forceinline__ float sigm(float x) {
  return 1.0f / (1.0f + __expf(-x));
}

// ---------------------------------------------------------------------------
// Small utility kernels
// ---------------------------------------------------------------------------
__global__ void zero_u32_kernel(uint32_t* p, int n) {
  int i = blockIdx.x * 256 + threadIdx.x;
  if (i < n) p[i] = 0u;
}

// Wcat[n][k] : k<E -> Wih[n][k]; E<=k<EPAD -> 0; k>=EPAD -> Whh[n][k-EPAD]
__global__ void build_wcat_kernel(const float* __restrict__ Wih,
                                  const float* __restrict__ Whh,
                                  __hip_bfloat16* __restrict__ Wcat) {
  int i = blockIdx.x * 256 + threadIdx.x;
  if (i >= G4H * KPAD) return;
  int n = i / KPAD, k = i % KPAD;
  float v = 0.0f;
  if (k < EE)        v = Wih[(size_t)n * EE + k];
  else if (k >= EPAD) v = Whh[(size_t)n * HH + (k - EPAD)];
  Wcat[i] = __float2bfloat16(v);
}

__global__ void bias_sum_kernel(const float* a, const float* b, float* o, int n) {
  int i = blockIdx.x * 256 + threadIdx.x;
  if (i < n) o[i] = a[i] + b[i];
}

__global__ void cast_bf16_kernel(const float* __restrict__ in,
                                 __hip_bfloat16* __restrict__ out, int n) {
  int i = blockIdx.x * 256 + threadIdx.x;
  if (i < n) out[i] = __float2bfloat16(in[i]);
}

__global__ void combine_desc_kernel(const int* __restrict__ anc, const int* __restrict__ alen,
                                    const int* __restrict__ neg, const int* __restrict__ nlen,
                                    int* __restrict__ ids, int* __restrict__ lens) {
  int i = blockIdx.x * 256 + threadIdx.x;
  if (i < BD * TDESC) {
    int b = i / TDESC, t = i % TDESC;
    ids[i] = (b < BB) ? anc[b * TDESC + t] : neg[(b - BB) * TDESC + t];
  }
  if (i < BD) lens[i] = (i < BB) ? alen[i] : nlen[i - BB];
}

// ---------------------------------------------------------------------------
// One-shot embedding gather -> padded bf16 rows: X[row*EPAD + k].
// ---------------------------------------------------------------------------
__global__ void embed_kernel(const int* __restrict__ ids,
                             const float* __restrict__ emb,
                             __hip_bfloat16* __restrict__ X, int rows) {
  int i = blockIdx.x * 256 + threadIdx.x;
  if (i >= rows * EPAD) return;
  int r = i / EPAD, k = i % EPAD;
  float v = 0.0f;
  if (k < EE) v = emb[(size_t)ids[r] * EE + k];
  X[i] = __float2bfloat16(v);
}

// ---------------------------------------------------------------------------
// Fused LSTM gate GEMM:  gates = [x_t | h_prev] @ Wcat^T + (bih+bhh)
//   LDS-free: A fragments loaded straight from global (rows shared via WGP$).
//   Each wave computes two 16x16 tiles (one A frag feeds two WMMAs).
//   grid = (Batch/16, 2048/256), block = 256 (8 waves).
// ---------------------------------------------------------------------------
__global__ __launch_bounds__(256) void lstm_gate_gemm_kernel(
    const __hip_bfloat16* __restrict__ X, int T, int t,
    const __hip_bfloat16* __restrict__ Wcat,
    const __hip_bfloat16* __restrict__ hprev,
    const float* __restrict__ bias,
    float* __restrict__ gates) {
  int tid = threadIdx.x;
  int wave = tid >> 5, lane = tid & 31, half = lane >> 4, lm = lane & 15;
  int m0 = blockIdx.x * 16;
  int n0 = blockIdx.y * 256 + wave * 32;   // two tiles: n0, n0+16

  const __hip_bfloat16* arow_x = X + ((size_t)(m0 + lm) * T + t) * EPAD;
  const __hip_bfloat16* arow_h = hprev + (size_t)(m0 + lm) * HH;
  const __hip_bfloat16* brow0  = Wcat + (size_t)(n0 + lm) * KPAD + half * 16;
  const __hip_bfloat16* brow1  = Wcat + (size_t)(n0 + 16 + lm) * KPAD + half * 16;

  v8f acc0 = {0.f, 0.f, 0.f, 0.f, 0.f, 0.f, 0.f, 0.f};
  v8f acc1 = {0.f, 0.f, 0.f, 0.f, 0.f, 0.f, 0.f, 0.f};

  // ---- x_t part: K in [0, EPAD) ----
#pragma unroll
  for (int k0 = 0; k0 < EPAD; k0 += 32) {
    v16bf a  = load_afrag(arow_x, k0, half);
    v16bf b0 = *(const v16bf*)(brow0 + k0);
    v16bf b1 = *(const v16bf*)(brow1 + k0);
    acc0 = wmma_bf16(a, b0, acc0);
    acc1 = wmma_bf16(a, b1, acc1);
  }
  // ---- h_prev part: K in [EPAD, KPAD) ----
#pragma unroll
  for (int k0 = 0; k0 < HH; k0 += 32) {
    v16bf a  = load_afrag(arow_h, k0, half);
    v16bf b0 = *(const v16bf*)(brow0 + EPAD + k0);
    v16bf b1 = *(const v16bf*)(brow1 + EPAD + k0);
    acc0 = wmma_bf16(a, b0, acc0);
    acc1 = wmma_bf16(a, b1, acc1);
  }

#pragma unroll
  for (int tile = 0; tile < 2; ++tile) {
    int n = n0 + tile * 16 + lm;
    float bn = bias[n];
    const v8f& acc = tile ? acc1 : acc0;
#pragma unroll
    for (int v = 0; v < 8; ++v) {
      int m = m0 + half * 8 + v;
      gates[(size_t)m * G4H + n] = acc[v] + bn;
    }
  }
}

// ---------------------------------------------------------------------------
// Pointwise LSTM cell update.
// ---------------------------------------------------------------------------
__global__ void lstm_cell_kernel(const float* __restrict__ gates,
                                 float* __restrict__ c,
                                 __hip_bfloat16* __restrict__ h_bf,
                                 __hip_bfloat16* __restrict__ feat,   // or null
                                 float* __restrict__ finals,          // or null
                                 const int* __restrict__ lens,        // or null
                                 int Batch, int t, int T) {
  int idx = blockIdx.x * 256 + threadIdx.x;
  if (idx >= Batch * HH) return;
  int b = idx / HH, h = idx % HH;
  const float* g = gates + (size_t)b * G4H;
  float gi = g[h], gf = g[HH + h], gg = g[2 * HH + h], go = g[3 * HH + h];
  float cn = sigm(gf) * c[idx] + sigm(gi) * tanhf(gg);
  float hn = sigm(go) * tanhf(cn);
  c[idx] = cn;
  h_bf[idx] = __float2bfloat16(hn);
  float th = tanhf(hn);
  if (feat) feat[((size_t)b * T + t) * HH + h] = __float2bfloat16(th);
  if (finals && t == lens[b] - 1) finals[idx] = th;
}

// ---------------------------------------------------------------------------
// Fused attention score:  scores[bt] = sum_n tanh(feat[bt]@W[n]^T + b[n])*sW[n] + sB
//   LDS-free GEMM; one small LDS array only for the cross-lane score reduce.
//   block handles 16 rows x 512 cols; 8 waves x 4 N-tiles each.
// ---------------------------------------------------------------------------
__global__ __launch_bounds__(256) void attn_score_kernel(
    const __hip_bfloat16* __restrict__ feat,
    const __hip_bfloat16* __restrict__ Wb,
    const float* __restrict__ attn_b,
    const float* __restrict__ sW, const float* __restrict__ sB,
    float* __restrict__ scores) {
  __shared__ float sS[16];
  int tid = threadIdx.x;
  int wave = tid >> 5, lane = tid & 31, half = lane >> 4, lm = lane & 15;
  int m0 = blockIdx.x * 16;

  const __hip_bfloat16* arow = feat + (size_t)(m0 + lm) * HH;

  v8f acc[4];
#pragma unroll
  for (int j = 0; j < 4; ++j) acc[j] = (v8f){0.f, 0.f, 0.f, 0.f, 0.f, 0.f, 0.f, 0.f};

#pragma unroll 4
  for (int k0 = 0; k0 < HH; k0 += 32) {
    v16bf a = load_afrag(arow, k0, half);
#pragma unroll
    for (int j = 0; j < 4; ++j) {
      int n0 = (wave * 4 + j) * 16;
      v16bf bf = *(const v16bf*)(Wb + (size_t)(n0 + lm) * HH + k0 + half * 16);
      acc[j] = wmma_bf16(a, bf, acc[j]);
    }
  }

  if (tid < 16) sS[tid] = 0.0f;
  __syncthreads();
#pragma unroll
  for (int j = 0; j < 4; ++j) {
    int n = (wave * 4 + j) * 16 + lm;
    float bn = attn_b[n], wn = sW[n];
#pragma unroll
    for (int v = 0; v < 8; ++v) {
      float u = tanhf(acc[j][v] + bn);
      atomicAdd(&sS[half * 8 + v], u * wn);
    }
  }
  __syncthreads();
  if (tid < 16) scores[m0 + tid] = sS[tid] + sB[0];
}

// ---------------------------------------------------------------------------
// Masked softmax over T=200 per batch row.
// ---------------------------------------------------------------------------
__global__ void softmax_mask_kernel(const float* __restrict__ scores,
                                    const int* __restrict__ lens,
                                    float* __restrict__ w) {
  __shared__ float sh[256];
  int b = blockIdx.x, t = threadIdx.x;
  float s = -1e9f;
  if (t < TCODE && t < lens[b]) s = scores[b * TCODE + t];
  sh[t] = s;
  __syncthreads();
  for (int st = 128; st > 0; st >>= 1) {
    if (t < st) sh[t] = fmaxf(sh[t], sh[t + st]);
    __syncthreads();
  }
  float m = sh[0];
  __syncthreads();
  float e = (t < TCODE) ? __expf(s - m) : 0.0f;
  sh[t] = e;
  __syncthreads();
  for (int st = 128; st > 0; st >>= 1) {
    if (t < st) sh[t] += sh[t + st];
    __syncthreads();
  }
  if (t < TCODE) w[b * TCODE + t] = e / sh[0];
}

// pooled[b,h] = sum_t w[b,t] * feat[b,t,h]
__global__ void pool_kernel(const float* __restrict__ w,
                            const __hip_bfloat16* __restrict__ feat,
                            float* __restrict__ pooled) {
  int b = blockIdx.x;
  for (int h = threadIdx.x; h < HH; h += 256) {
    float acc = 0.0f;
    for (int t = 0; t < TCODE; ++t)
      acc += w[b * TCODE + t] *
             __bfloat162float(feat[((size_t)b * TCODE + t) * HH + h]);
    pooled[(size_t)b * HH + h] = acc;
  }
}

// ---------------------------------------------------------------------------
// Generic WMMA GEMM + bias + tanh:  out = tanh(A @ W^T + bias). LDS-free.
//   A: (M x Ka) bf16 row-major, W: (N x Ka) bf16 n-major.
//   grid = (M/16, N/128), block = 256.
// ---------------------------------------------------------------------------
__global__ __launch_bounds__(256) void gemm_tanh_kernel(
    const __hip_bfloat16* __restrict__ A,
    const __hip_bfloat16* __restrict__ W,
    const float* __restrict__ bias,
    float* __restrict__ outF,            // may be null
    __hip_bfloat16* __restrict__ outB,   // may be null
    int Ka, int Nld) {
  int tid = threadIdx.x;
  int wave = tid >> 5, lane = tid & 31, half = lane >> 4, lm = lane & 15;
  int m0 = blockIdx.x * 16;
  int n0 = blockIdx.y * 128 + wave * 16;

  const __hip_bfloat16* arow = A + (size_t)(m0 + lm) * Ka;
  const __hip_bfloat16* brow = W + (size_t)(n0 + lm) * Ka + half * 16;

  v8f acc = {0.f, 0.f, 0.f, 0.f, 0.f, 0.f, 0.f, 0.f};
#pragma unroll 4
  for (int k0 = 0; k0 < Ka; k0 += 32) {
    v16bf a  = load_afrag(arow, k0, half);
    v16bf bf = *(const v16bf*)(brow + k0);
    acc = wmma_bf16(a, bf, acc);
  }
  int n = n0 + lm;
  float bn = bias[n];
#pragma unroll
  for (int v = 0; v < 8; ++v) {
    int m = m0 + half * 8 + v;
    float val = tanhf(acc[v] + bn);
    if (outF) outF[(size_t)m * Nld + n] = val;
    if (outB) outB[(size_t)m * Nld + n] = __float2bfloat16(val);
  }
}

// ---------------------------------------------------------------------------
// Triplet loss: mean_b max(MARGIN - cos(code,anc) + cos(code,neg), 1e-6)
// ---------------------------------------------------------------------------
__global__ void loss_kernel(const float* __restrict__ code_vec,
                            const float* __restrict__ finals,
                            float* __restrict__ out) {
  __shared__ float red[256];
  int b = threadIdx.x;
  float da = 0, dn = 0, nc = 0, na = 0, nn = 0;
  for (int j = 0; j < HH; ++j) {
    float c = code_vec[(size_t)b * HH + j];
    float a = finals[(size_t)b * HH + j];
    float g = finals[(size_t)(BB + b) * HH + j];
    da += c * a; dn += c * g;
    nc += c * c; na += a * a; nn += g * g;
  }
  float cosa = da / fmaxf(sqrtf(nc) * sqrtf(na), 1e-8f);
  float cosn = dn / fmaxf(sqrtf(nc) * sqrtf(nn), 1e-8f);
  red[b] = fmaxf(0.6f - cosa + cosn, 1e-6f);
  __syncthreads();
  for (int st = 128; st > 0; st >>= 1) {
    if (b < st) red[b] += red[b + st];
    __syncthreads();
  }
  if (b == 0) out[0] = red[0] / (float)BB;
}

// ---------------------------------------------------------------------------
// Host orchestration
// ---------------------------------------------------------------------------
extern "C" void kernel_launch(void* const* d_in, const int* in_sizes, int n_in,
                              void* d_out, int out_size, void* d_ws, size_t ws_size,
                              hipStream_t stream) {
  (void)in_sizes; (void)n_in; (void)out_size; (void)ws_size;

  const int*   tokens   = (const int*)d_in[0];
  const int*   tok_len  = (const int*)d_in[1];
  const int*   danc     = (const int*)d_in[2];
  const int*   danc_len = (const int*)d_in[3];
  const int*   dneg     = (const int*)d_in[4];
  const int*   dneg_len = (const int*)d_in[5];
  const float* tok_emb  = (const float*)d_in[6];
  const float* tok_Wih  = (const float*)d_in[7];
  const float* tok_Whh  = (const float*)d_in[8];
  const float* tok_bih  = (const float*)d_in[9];
  const float* tok_bhh  = (const float*)d_in[10];
  const float* desc_emb = (const float*)d_in[11];
  const float* desc_Wih = (const float*)d_in[12];
  const float* desc_Whh = (const float*)d_in[13];
  const float* desc_bih = (const float*)d_in[14];
  const float* desc_bhh = (const float*)d_in[15];
  const float* attn_W   = (const float*)d_in[16];
  const float* attn_b   = (const float*)d_in[17];
  const float* attnS_W  = (const float*)d_in[18];
  const float* attnS_b  = (const float*)d_in[19];
  const float* out_W1   = (const float*)d_in[20];
  const float* out_b1   = (const float*)d_in[21];
  const float* out_W2   = (const float*)d_in[22];
  const float* out_b2   = (const float*)d_in[23];
  float* out = (float*)d_out;

  char* ws = (char*)d_ws;
  size_t off = 0;
  auto alloc = [&](size_t bytes) -> char* {
    char* p = ws + off;
    off += (bytes + 255) & ~(size_t)255;
    return p;
  };

  __hip_bfloat16* wcat_c  = (__hip_bfloat16*)alloc((size_t)G4H * KPAD * 2);
  __hip_bfloat16* wcat_d  = (__hip_bfloat16*)alloc((size_t)G4H * KPAD * 2);
  float*          bias_c  = (float*)alloc(G4H * 4);
  float*          bias_d  = (float*)alloc(G4H * 4);
  __hip_bfloat16* wattn   = (__hip_bfloat16*)alloc((size_t)HH * HH * 2);
  __hip_bfloat16* w1b     = (__hip_bfloat16*)alloc((size_t)HH * HH * 2);
  __hip_bfloat16* w2b     = (__hip_bfloat16*)alloc((size_t)HH * HH * 2);
  __hip_bfloat16* h_code  = (__hip_bfloat16*)alloc((size_t)BB * HH * 2);
  float*          c_code  = (float*)alloc((size_t)BB * HH * 4);
  __hip_bfloat16* h_desc  = (__hip_bfloat16*)alloc((size_t)BD * HH * 2);
  float*          c_desc  = (float*)alloc((size_t)BD * HH * 4);
  float*          gates   = (float*)alloc((size_t)BD * G4H * 4);
  __hip_bfloat16* feat    = (__hip_bfloat16*)alloc((size_t)MROWS * HH * 2);
  __hip_bfloat16* x_code  = (__hip_bfloat16*)alloc((size_t)MROWS * EPAD * 2);
  __hip_bfloat16* x_desc  = (__hip_bfloat16*)alloc((size_t)BD * TDESC * EPAD * 2);
  float*          scores  = (float*)alloc((size_t)MROWS * 4);
  float*          attw    = (float*)alloc((size_t)BB * TCODE * 4);
  float*          pooled  = (float*)alloc((size_t)BB * HH * 4);
  __hip_bfloat16* pooledb = (__hip_bfloat16*)alloc((size_t)BB * HH * 2);
  __hip_bfloat16* mlp1b   = (__hip_bfloat16*)alloc((size_t)BB * HH * 2);
  float*          codevec = (float*)alloc((size_t)BB * HH * 4);
  float*          dfinal  = (float*)alloc((size_t)BD * HH * 4);
  int*            ids_d   = (int*)alloc((size_t)BD * TDESC * 4);
  int*            lens_d  = (int*)alloc((size_t)BD * 4);

  auto gridFor = [](int n) { return dim3((n + 255) / 256); };

  // ---- weight prep (bf16 copies, fused biases) ----
  build_wcat_kernel<<<gridFor(G4H * KPAD), 256, 0, stream>>>(tok_Wih, tok_Whh, wcat_c);
  build_wcat_kernel<<<gridFor(G4H * KPAD), 256, 0, stream>>>(desc_Wih, desc_Whh, wcat_d);
  bias_sum_kernel<<<gridFor(G4H), 256, 0, stream>>>(tok_bih, tok_bhh, bias_c, G4H);
  bias_sum_kernel<<<gridFor(G4H), 256, 0, stream>>>(desc_bih, desc_bhh, bias_d, G4H);
  cast_bf16_kernel<<<gridFor(HH * HH), 256, 0, stream>>>(attn_W, wattn, HH * HH);
  cast_bf16_kernel<<<gridFor(HH * HH), 256, 0, stream>>>(out_W1, w1b, HH * HH);
  cast_bf16_kernel<<<gridFor(HH * HH), 256, 0, stream>>>(out_W2, w2b, HH * HH);
  combine_desc_kernel<<<gridFor(BD * TDESC), 256, 0, stream>>>(danc, danc_len, dneg,
                                                               dneg_len, ids_d, lens_d);

  // ---- one-shot embedding gather (off the recurrent critical path) ----
  embed_kernel<<<gridFor(MROWS * EPAD), 256, 0, stream>>>(tokens, tok_emb, x_code, MROWS);
  embed_kernel<<<gridFor(BD * TDESC * EPAD), 256, 0, stream>>>(ids_d, desc_emb, x_desc,
                                                               BD * TDESC);

  // ---- zero initial LSTM state ----
  zero_u32_kernel<<<gridFor(BB * HH / 2), 256, 0, stream>>>((uint32_t*)h_code, BB * HH / 2);
  zero_u32_kernel<<<gridFor(BB * HH), 256, 0, stream>>>((uint32_t*)c_code, BB * HH);
  zero_u32_kernel<<<gridFor(BD * HH / 2), 256, 0, stream>>>((uint32_t*)h_desc, BD * HH / 2);
  zero_u32_kernel<<<gridFor(BD * HH), 256, 0, stream>>>((uint32_t*)c_desc, BD * HH);

  // ---- code encoder LSTM (B=256, T=200) ----
  {
    dim3 g(BB / 16, G4H / 256);
    for (int t = 0; t < TCODE; ++t) {
      lstm_gate_gemm_kernel<<<g, 256, 0, stream>>>(x_code, TCODE, t, wcat_c,
                                                   h_code, bias_c, gates);
      lstm_cell_kernel<<<gridFor(BB * HH), 256, 0, stream>>>(
          gates, c_code, h_code, feat, nullptr, nullptr, BB, t, TCODE);
    }
  }

  // ---- desc encoders (anchor+neg batched, B=512, T=50) ----
  {
    dim3 g(BD / 16, G4H / 256);
    for (int t = 0; t < TDESC; ++t) {
      lstm_gate_gemm_kernel<<<g, 256, 0, stream>>>(x_desc, TDESC, t, wcat_d,
                                                   h_desc, bias_d, gates);
      lstm_cell_kernel<<<gridFor(BD * HH), 256, 0, stream>>>(
          gates, c_desc, h_desc, nullptr, dfinal, lens_d, BD, t, TDESC);
    }
  }

  // ---- attention scores + masked softmax + pooling ----
  attn_score_kernel<<<dim3(MROWS / 16), 256, 0, stream>>>(feat, wattn, attn_b,
                                                          attnS_W, attnS_b, scores);
  softmax_mask_kernel<<<dim3(BB), 256, 0, stream>>>(scores, tok_len, attw);
  pool_kernel<<<dim3(BB), 256, 0, stream>>>(attw, feat, pooled);

  // ---- output MLP: code_vec = tanh(tanh(pooled@W1^T+b1)@W2^T+b2) ----
  cast_bf16_kernel<<<gridFor(BB * HH), 256, 0, stream>>>(pooled, pooledb, BB * HH);
  gemm_tanh_kernel<<<dim3(BB / 16, HH / 128), 256, 0, stream>>>(
      pooledb, w1b, out_b1, nullptr, mlp1b, HH, HH);
  gemm_tanh_kernel<<<dim3(BB / 16, HH / 128), 256, 0, stream>>>(
      mlp1b, w2b, out_b2, codevec, nullptr, HH, HH);

  // ---- triplet loss ----
  loss_kernel<<<dim3(1), 256, 0, stream>>>(codevec, dfinal, out);
}
